// gtaformerIteration_84121229459774
// MI455X (gfx1250) — compile-verified, matrix operand order checked by reading, not run on hardware
//
#include <hip/hip_runtime.h>
#include <math.h>

// ---- problem constants ----
#define LL_   512
#define CSEQ  256
#define CPAIR 128
#define NH    8
#define DD    32
#define NOUT  32
#define EPSV  1e-5f

typedef __attribute__((ext_vector_type(16))) _Float16 v16h;
typedef __attribute__((ext_vector_type(8)))  float    v8f;

// K index map for 16-bit A/B fragments (CDNA5 ISA 7.12.2):
// lane half hh selects K group; element e (0..15) -> K
__device__ __forceinline__ int kmap(int e, int hh) {
  return (e < 8) ? (8 * hh + e) : (8 + 8 * hh + e); // e>=8: 16 + 8*hh + (e-8)
}

// A fragment (16x32, f32 source row-major, lda)
__device__ __forceinline__ v16h load_a_f32(const float* __restrict__ A, int lda,
                                           int m_base, int k_base, int lane) {
  int m = m_base + (lane & 15);
  int hh = lane >> 4;
  v16h a;
#pragma unroll
  for (int e = 0; e < 16; ++e)
    a[e] = (_Float16)A[m * lda + k_base + kmap(e, hh)];
  return a;
}

// B fragment (32x16, f32 source row-major [k][n], ldb)
__device__ __forceinline__ v16h load_b_f32(const float* __restrict__ B, int ldb,
                                           int k_base, int n_base, int lane) {
  int n = n_base + (lane & 15);
  int hh = lane >> 4;
  v16h b;
#pragma unroll
  for (int e = 0; e < 16; ++e)
    b[e] = (_Float16)B[(k_base + kmap(e, hh)) * ldb + n];
  return b;
}

__device__ __forceinline__ float wave_sum(float v) {
#pragma unroll
  for (int m = 16; m >= 1; m >>= 1) v += __shfl_xor(v, m, 32);
  return v;
}
__device__ __forceinline__ float wave_max(float v) {
#pragma unroll
  for (int m = 16; m >= 1; m >>= 1) v = fmaxf(v, __shfl_xor(v, m, 32));
  return v;
}

// ---------- K1: pair layernorm + bias einsum (single pass over pair_act) ----------
// one wave per (i,j); bias stored [h][i][j]
__global__ void k_pair_bias(const float* __restrict__ pair,
                            const float* __restrict__ pn_g, const float* __restrict__ pn_b,
                            const float* __restrict__ W_pair, float* __restrict__ bias) {
  int lane = threadIdx.x & 31, wave = threadIdx.x >> 5;
  int p = blockIdx.x * 8 + wave; // 0..262143
  const float* row = pair + (size_t)p * CPAIR;
  float x[4], s = 0.f, s2 = 0.f;
#pragma unroll
  for (int e = 0; e < 4; ++e) { x[e] = row[lane * 4 + e]; s += x[e]; s2 += x[e] * x[e]; }
  s = wave_sum(s); s2 = wave_sum(s2);
  float mean = s * (1.f / CPAIR);
  float var  = s2 * (1.f / CPAIR) - mean * mean;
  float inv  = rsqrtf(var + EPSV);
  float acc[8];
#pragma unroll
  for (int h = 0; h < 8; ++h) acc[h] = 0.f;
#pragma unroll
  for (int e = 0; e < 4; ++e) {
    int c = lane * 4 + e;
    float xn = (x[e] - mean) * inv * pn_g[c] + pn_b[c];
#pragma unroll
    for (int h = 0; h < 8; ++h) acc[h] += xn * W_pair[c * 8 + h];
  }
#pragma unroll
  for (int h = 0; h < 8; ++h) acc[h] = wave_sum(acc[h]);
  if (lane < 8) {
    int i = p >> 9, j = p & 511;
    bias[(size_t)lane * (LL_ * LL_) + i * LL_ + j] = acc[lane];
  }
}

// ---------- K2: both seq layernorms ----------
__global__ void k_seq_ln(const float* __restrict__ seq,
                         const float* __restrict__ sn_g, const float* __restrict__ sn_b,
                         const float* __restrict__ on_g, const float* __restrict__ on_b,
                         float* __restrict__ seq_n, float* __restrict__ a_n) {
  int lane = threadIdx.x & 31, wave = threadIdx.x >> 5;
  int l = blockIdx.x * 8 + wave;
  const float* row = seq + l * CSEQ;
  float x[8], s = 0.f, s2 = 0.f;
#pragma unroll
  for (int e = 0; e < 8; ++e) {
    int c = e * 32 + lane;
    x[e] = row[c]; s += x[e]; s2 += x[e] * x[e];
  }
  s = wave_sum(s); s2 = wave_sum(s2);
  float mean = s * (1.f / CSEQ);
  float var  = s2 * (1.f / CSEQ) - mean * mean;
  float inv  = rsqrtf(var + EPSV);
#pragma unroll
  for (int e = 0; e < 8; ++e) {
    int c = e * 32 + lane;
    float xn = (x[e] - mean) * inv;
    seq_n[l * CSEQ + c] = xn * sn_g[c] + sn_b[c];
    a_n[l * CSEQ + c]   = xn * on_g[c] + on_b[c];
  }
}

// ---------- K3: la / ra ----------
__global__ void k_lr(const float* __restrict__ a_n,
                     const float* __restrict__ W_l, const float* __restrict__ b_l,
                     const float* __restrict__ W_r, const float* __restrict__ b_r,
                     const float* __restrict__ mask,
                     float* __restrict__ la, float* __restrict__ ra) {
  int lane = threadIdx.x & 31, wave = threadIdx.x >> 5;
  int l = blockIdx.x * 8 + wave;
  const float* row = a_n + l * CSEQ;
  float al = 0.f, ar = 0.f;
  for (int c = 0; c < CSEQ; ++c) {
    float av = row[c];
    al += av * W_l[c * NOUT + lane];
    ar += av * W_r[c * NOUT + lane];
  }
  float m = mask[l];
  la[l * NOUT + lane] = m * (al + b_l[lane]);
  ra[l * NOUT + lane] = m * (ar + b_r[lane]);
}

// ---------- K4: WMMA projection: seq_n @ W_in (q0/k0/v0) and seq_n @ W_gate (sigmoid) ----------
__global__ void k_proj(const float* __restrict__ seq_n, const float* __restrict__ W_in,
                       const float* __restrict__ W_gate, const float* __restrict__ b_gate,
                       float* __restrict__ q0, float* __restrict__ k0, float* __restrict__ v0,
                       float* __restrict__ gate) {
  int lane = threadIdx.x & 31, wave = threadIdx.x >> 5;
  int mt = blockIdx.x;                 // 0..31 (rows of 16)
  int nt = blockIdx.y * 8 + wave;      // 0..63: 0..47 -> W_in, 48..63 -> W_gate
  v8f c = {};
  if (nt < 48) {
#pragma unroll
    for (int kc = 0; kc < 8; ++kc) {
      v16h a = load_a_f32(seq_n, CSEQ, mt * 16, kc * 32, lane);
      v16h b = load_b_f32(W_in, 3 * CSEQ, kc * 32, nt * 16, lane);
      c = __builtin_amdgcn_wmma_f32_16x16x32_f16(false, a, false, b, (short)0, c, false, false);
    }
    int n = lane & 15, hh = lane >> 4;
    int col = nt * 16 + n;
    int h = col / 96, rem = col % 96, s = rem / 32, d = rem % 32;
    float* dst = (s == 0) ? q0 : (s == 1) ? k0 : v0;
#pragma unroll
    for (int r = 0; r < 8; ++r) {
      int l = mt * 16 + r + 8 * hh;
      dst[h * (LL_ * DD) + l * DD + d] = c[r];
    }
  } else {
    int gt = nt - 48;
#pragma unroll
    for (int kc = 0; kc < 8; ++kc) {
      v16h a = load_a_f32(seq_n, CSEQ, mt * 16, kc * 32, lane);
      v16h b = load_b_f32(W_gate, CSEQ, kc * 32, gt * 16, lane);
      c = __builtin_amdgcn_wmma_f32_16x16x32_f16(false, a, false, b, (short)0, c, false, false);
    }
    int n = lane & 15, hh = lane >> 4;
    int col = gt * 16 + n;
#pragma unroll
    for (int r = 0; r < 8; ++r) {
      int l = mt * 16 + r + 8 * hh;
      float val = c[r] + b_gate[col];
      gate[l * CSEQ + col] = 1.f / (1.f + expf(-val));
    }
  }
}

// ---------- K5: inception depthwise conv (identity heads 0-1; k=3/5/7 for head pairs) ----------
__global__ void k_conv(const float* __restrict__ in, float* __restrict__ out,
                       const float* __restrict__ w3, const float* __restrict__ b3,
                       const float* __restrict__ w5, const float* __restrict__ b5,
                       const float* __restrict__ w7, const float* __restrict__ b7,
                       float scale) {
  int idx = blockIdx.x * 256 + threadIdx.x; // h*16384 + l*32 + d
  int h = idx >> 14;
  int l = (idx >> 5) & 511;
  int d = idx & 31;
  float val;
  if (h < 2) {
    val = in[idx];
  } else {
    int g = (h - 2) >> 1;                  // 0,1,2 -> k=3,5,7
    const float* w  = (g == 0) ? w3 : (g == 1) ? w5 : w7;
    const float* bb = (g == 0) ? b3 : (g == 1) ? b5 : b7;
    int ksz = 3 + 2 * g;
    int pad = g + 1;
    float acc = bb[d];
    const float* base = in + h * (LL_ * DD) + d;
    for (int kk = 0; kk < ksz; ++kk) {
      int ll = l + kk - pad;
      if (ll >= 0 && ll < LL_) acc += base[ll * DD] * w[d * ksz + kk];
    }
    val = acc;
  }
  out[idx] = val * scale;
}

// ---------- K6: attention: WMMA QK^T (+bias,mask) -> LDS softmax -> P.V, fused gate ----------
__global__ void k_attn(const float* __restrict__ q, const float* __restrict__ k,
                       const float* __restrict__ v, const float* __restrict__ bias,
                       const float* __restrict__ mask, const float* __restrict__ gate,
                       float* __restrict__ gwa) {
  __shared__ float P[16 * LL_]; // 32 KB
  int lane = threadIdx.x & 31, wave = threadIdx.x >> 5;
  int h  = blockIdx.x >> 5;    // 0..7
  int qt = blockIdx.x & 31;    // 0..31
  const float* qh = q + h * (LL_ * DD);
  const float* kh = k + h * (LL_ * DD);
  const float* vh = v + h * (LL_ * DD);
  const float* bh = bias + (size_t)h * (LL_ * LL_);

  // phase 1: 16x512 logits, 4 ki-tiles per wave
  v16h a = load_a_f32(qh, DD, qt * 16, 0, lane);
  int hh = lane >> 4, n = lane & 15;
  for (int t = 0; t < 4; ++t) {
    int kt = wave * 4 + t;
    int ki = kt * 16 + n;
    v16h b;
#pragma unroll
    for (int e = 0; e < 16; ++e) b[e] = (_Float16)kh[ki * DD + kmap(e, hh)];
    v8f c = {};
    c = __builtin_amdgcn_wmma_f32_16x16x32_f16(false, a, false, b, (short)0, c, false, false);
    float km = mask[ki];
#pragma unroll
    for (int r = 0; r < 8; ++r) {
      int row = r + 8 * hh;
      int qi = qt * 16 + row;
      float lg = c[r] + bh[qi * LL_ + ki];
      P[row * LL_ + kt * 16 + n] = (km > 0.f) ? lg : -3.4028235e38f;
    }
  }
  __syncthreads();

  // phase 2: softmax (2 rows per wave)
  for (int rr = 0; rr < 2; ++rr) {
    int row = wave * 2 + rr;
    float mx = -3.4028235e38f;
#pragma unroll
    for (int e = 0; e < 16; ++e) mx = fmaxf(mx, P[row * LL_ + e * 32 + lane]);
    mx = wave_max(mx);
    float sm = 0.f;
#pragma unroll
    for (int e = 0; e < 16; ++e) {
      float w = expf(P[row * LL_ + e * 32 + lane] - mx);
      P[row * LL_ + e * 32 + lane] = w;
      sm += w;
    }
    sm = wave_sum(sm);
    float invs = 1.f / sm;
#pragma unroll
    for (int e = 0; e < 16; ++e) P[row * LL_ + e * 32 + lane] *= invs;
  }
  __syncthreads();

  // phase 3: wa = P.V, fused with gate -> gwa[l][h*32+d]
  for (int rr = 0; rr < 2; ++rr) {
    int row = wave * 2 + rr;
    int qi = qt * 16 + row;
    float acc = 0.f;
    for (int ki = 0; ki < LL_; ++ki)
      acc += P[row * LL_ + ki] * vh[ki * DD + lane];
    int cch = h * DD + lane;
    gwa[qi * CSEQ + cch] = acc * gate[qi * CSEQ + cch];
  }
}

// ---------- K7: seq_out = gwa @ W_out + b_out (WMMA) ----------
__global__ void k_outproj(const float* __restrict__ gwa, const float* __restrict__ W_out,
                          const float* __restrict__ b_out, float* __restrict__ seq_out) {
  int lane = threadIdx.x & 31, wave = threadIdx.x >> 5;
  int mt = blockIdx.x;
  int nt = blockIdx.y * 8 + wave; // 0..15
  v8f c = {};
#pragma unroll
  for (int kc = 0; kc < 8; ++kc) {
    v16h a = load_a_f32(gwa, CSEQ, mt * 16, kc * 32, lane);
    v16h b = load_b_f32(W_out, CSEQ, kc * 32, nt * 16, lane);
    c = __builtin_amdgcn_wmma_f32_16x16x32_f16(false, a, false, b, (short)0, c, false, false);
  }
  int n = lane & 15, hh = lane >> 4;
  int col = nt * 16 + n;
#pragma unroll
  for (int r = 0; r < 8; ++r) {
    int l = mt * 16 + r + 8 * hh;
    seq_out[l * CSEQ + col] = c[r] + b_out[col];
  }
}

// ---------- K8: pair_out[i][j][:] = [la_j*ra_i ; la_j-ra_i](64) @ W_o + b_o (WMMA, streamed) ----------
__global__ void k_pair_out(const float* __restrict__ la, const float* __restrict__ ra,
                           const float* __restrict__ W_o, const float* __restrict__ b_o,
                           float* __restrict__ pair_out) {
  int lane = threadIdx.x & 31, wave = threadIdx.x >> 5;
  int i  = blockIdx.x;  // 0..511
  int nt = wave;        // 8 waves -> 128 cols
  int hh = lane >> 4, n = lane & 15;
  v16h b0 = load_b_f32(W_o, CPAIR, 0,  nt * 16, lane); // K 0..31 (prod rows)
  v16h b1 = load_b_f32(W_o, CPAIR, 32, nt * 16, lane); // K 32..63 (diff rows)
  float rv[16];
#pragma unroll
  for (int e = 0; e < 16; ++e) rv[e] = ra[i * NOUT + kmap(e, hh)];
  int col = nt * 16 + n;
  float bo = b_o[col];
  float* dst = pair_out + (size_t)i * (LL_ * CPAIR);
  for (int jt = 0; jt < 32; ++jt) {
    int j = jt * 16 + n;
    v16h a0, a1;
#pragma unroll
    for (int e = 0; e < 16; ++e) {
      float lv = la[j * NOUT + kmap(e, hh)];
      a0[e] = (_Float16)(lv * rv[e]);
      a1[e] = (_Float16)(lv - rv[e]);
    }
    v8f c = {};
    c = __builtin_amdgcn_wmma_f32_16x16x32_f16(false, a0, false, b0, (short)0, c, false, false);
    c = __builtin_amdgcn_wmma_f32_16x16x32_f16(false, a1, false, b1, (short)0, c, false, false);
#pragma unroll
    for (int r = 0; r < 8; ++r) {
      int jr = jt * 16 + r + 8 * hh;
      dst[jr * CPAIR + col] = c[r] + bo;
    }
  }
}

extern "C" void kernel_launch(void* const* d_in, const int* in_sizes, int n_in,
                              void* d_out, int out_size, void* d_ws, size_t ws_size,
                              hipStream_t stream) {
  (void)in_sizes; (void)n_in; (void)out_size; (void)ws_size;
  const float* seq_act  = (const float*)d_in[0];
  const float* pair_act = (const float*)d_in[1];
  const float* mask     = (const float*)d_in[2];
  const float* sn_g = (const float*)d_in[3];
  const float* sn_b = (const float*)d_in[4];
  const float* pn_g = (const float*)d_in[5];
  const float* pn_b = (const float*)d_in[6];
  const float* W_pair = (const float*)d_in[7];
  const float* W_in   = (const float*)d_in[8];
  const float* W_gate = (const float*)d_in[9];
  const float* b_gate = (const float*)d_in[10];
  const float* W_out  = (const float*)d_in[11];
  const float* b_out  = (const float*)d_in[12];
  const float* on_g = (const float*)d_in[13];
  const float* on_b = (const float*)d_in[14];
  const float* W_l = (const float*)d_in[15];
  const float* b_l = (const float*)d_in[16];
  const float* W_r = (const float*)d_in[17];
  const float* b_r = (const float*)d_in[18];
  const float* W_o = (const float*)d_in[19];
  const float* b_o = (const float*)d_in[20];
  // conv params: q (21..26), k (27..32), v (33..38); each: w[0..2] then b[0..2]
  const float* cq_w[3] = { (const float*)d_in[21], (const float*)d_in[22], (const float*)d_in[23] };
  const float* cq_b[3] = { (const float*)d_in[24], (const float*)d_in[25], (const float*)d_in[26] };
  const float* ck_w[3] = { (const float*)d_in[27], (const float*)d_in[28], (const float*)d_in[29] };
  const float* ck_b[3] = { (const float*)d_in[30], (const float*)d_in[31], (const float*)d_in[32] };
  const float* cv_w[3] = { (const float*)d_in[33], (const float*)d_in[34], (const float*)d_in[35] };
  const float* cv_b[3] = { (const float*)d_in[36], (const float*)d_in[37], (const float*)d_in[38] };

  float* ws = (float*)d_ws;
  float* seq_n = ws;                 // 131072
  float* a_n   = ws + 131072;        // 131072
  float* gate  = ws + 262144;        // 131072
  float* q0    = ws + 393216;        // 131072  [h][l][d]
  float* k0    = ws + 524288;
  float* v0    = ws + 655360;
  float* qf    = ws + 786432;        // post-conv
  float* kf    = ws + 917504;
  float* vf    = ws + 1048576;
  float* la    = ws + 1179648;       // 16384
  float* ra    = ws + 1196032;       // 16384
  float* gwa   = ws + 1212416;       // 131072
  float* bias  = ws + 1343488;       // 8*512*512 = 2097152

  float* seq_out  = (float*)d_out;                 // 512*256
  float* pair_out = (float*)d_out + LL_ * CSEQ;    // 512*512*128

  k_pair_bias<<<32768, 256, 0, stream>>>(pair_act, pn_g, pn_b, W_pair, bias);
  k_seq_ln<<<64, 256, 0, stream>>>(seq_act, sn_g, sn_b, on_g, on_b, seq_n, a_n);
  k_lr<<<64, 256, 0, stream>>>(a_n, W_l, b_l, W_r, b_r, mask, la, ra);
  k_proj<<<dim3(32, 8), 256, 0, stream>>>(seq_n, W_in, W_gate, b_gate, q0, k0, v0, gate);
  const float qscale = 0.17677669529663687f; // 32^-0.5
  k_conv<<<512, 256, 0, stream>>>(q0, qf, cq_w[0], cq_b[0], cq_w[1], cq_b[1], cq_w[2], cq_b[2], qscale);
  k_conv<<<512, 256, 0, stream>>>(k0, kf, ck_w[0], ck_b[0], ck_w[1], ck_b[1], ck_w[2], ck_b[2], 1.0f);
  k_conv<<<512, 256, 0, stream>>>(v0, vf, cv_w[0], cv_b[0], cv_w[1], cv_b[1], cv_w[2], cv_b[2], 1.0f);
  k_attn<<<256, 256, 0, stream>>>(qf, kf, vf, bias, mask, gate, gwa);
  k_outproj<<<dim3(32, 2), 256, 0, stream>>>(gwa, W_out, b_out, seq_out);
  k_pair_out<<<512, 256, 0, stream>>>(la, ra, W_o, b_o, pair_out);
}
